// UTDGraphNetDenoiseV2_6176162972394
// MI455X (gfx1250) — matrix operand order repaired and checked by reading, not compiled, not run on hardware
//
#include <hip/hip_runtime.h>
#include <hip/hip_bf16.h>
#include <stddef.h>

// ---------------------------------------------------------------------------
// Problem constants (from the reference)
// ---------------------------------------------------------------------------
#define NN       10000
#define EE       320000
#define IN_DIM   256
#define HH       256
#define OUT_DIM  128
#define STEPS    3
#define TAU_THRESH 0.05f

typedef __attribute__((ext_vector_type(16))) __bf16 v16bf;
typedef __attribute__((ext_vector_type(8)))  float  v8f;

union FragBF { unsigned int u[8]; v16bf v; };

static __device__ __forceinline__ unsigned short f32_to_bf16_bits(float f) {
    unsigned int u = __float_as_uint(f);
    unsigned int r = (u + 0x7FFFu + ((u >> 16) & 1u)) >> 16;   // RNE
    return (unsigned short)r;
}

static __device__ __forceinline__ float sigmoidf(float x) {
    return 1.0f / (1.0f + __expf(-x));
}

// ---------------------------------------------------------------------------
// Zero fill (graph-capture-safe memset)
// ---------------------------------------------------------------------------
__global__ void zero_f32_kernel(float* __restrict__ p, int n) {
    int i = blockIdx.x * blockDim.x + threadIdx.x;
    int stride = gridDim.x * blockDim.x;
    for (; i < n; i += stride) p[i] = 0.0f;
}

// ---------------------------------------------------------------------------
// f32 -> bf16 conversion
// ---------------------------------------------------------------------------
__global__ void cvt_f32_bf16_kernel(const float* __restrict__ src,
                                    unsigned short* __restrict__ dst, int n) {
    int i = blockIdx.x * blockDim.x + threadIdx.x;
    int stride = gridDim.x * blockDim.x;
    for (; i < n; i += stride) dst[i] = f32_to_bf16_bits(src[i]);
}

// ---------------------------------------------------------------------------
// WMMA bf16 GEMM:  C[M,N] = act( A[M,K] * W[N,K]^T + bias[N] )
//   A, W are bf16 row-major; C f32; optional bf16 shadow of C.
//   Block = 256 threads = 8 waves; block tile 32x128; wave tile 16x32
//   (two 16x16 accumulators, A fragment reused); K step 32; LDS ping/pong
//   double-buffering -> one barrier per K iteration.
// ---------------------------------------------------------------------------
#define GT_M 32
#define GT_N 128
#define GT_K 32
#define LDS_STRIDE 40   // 32 + 8 pad shorts = 80 B rows (16B-aligned)

__global__ __launch_bounds__(256)
void gemm_bf16_wmma_kernel(const unsigned short* __restrict__ A,
                           const unsigned short* __restrict__ W,
                           const float* __restrict__ bias,
                           float* __restrict__ C,
                           unsigned short* __restrict__ Cbf,  // may be null
                           int M, int N, int K, int relu) {
    __shared__ unsigned short Alds[2][GT_M][LDS_STRIDE];
    __shared__ unsigned short Wlds[2][GT_N][LDS_STRIDE];

    const int tid  = threadIdx.x;
    const int lane = tid & 31;
    const int wave = tid >> 5;
    const int m0   = blockIdx.x * GT_M;
    const int n0   = blockIdx.y * GT_N;

    // wave tile position: 2 row-tiles x 4 col-pairs
    const int wrow = (wave >> 2) * 16;     // 0 or 16
    const int wcol = (wave & 3)  * 32;     // 0,32,64,96

    // staging maps (computed once)
    const int ar  = tid >> 3;              // A row 0..31
    const int akk = (tid & 7) << 2;        // A k offset 0,4,...,28 (b64)
    const int wr0 = tid >> 2;              // W rows: tid and tid+256 groups
    const int wk0 = (tid & 3) << 3;        // W k offset 0,8,16,24 (b128)
    const int wr1 = (tid + 256) >> 2;
    const int wk1 = ((tid + 256) & 3) << 3;

    v8f acc0 = {};
    v8f acc1 = {};

    const int l16 = lane & 15;
    const int kbA = (lane < 16) ? 0 : 16;  // A frag K base (contiguous k = 16h + s)
    const int kbB = (lane < 16) ? 0 : 16;  // B frag K base

    // ---- stage first K chunk into buffer 0 ----
    {
        uint2 av = make_uint2(0u, 0u);
        if (m0 + ar < M)
            av = *(const uint2*)(A + (size_t)(m0 + ar) * K + akk);
        *(uint2*)&Alds[0][ar][akk] = av;
        *(uint4*)&Wlds[0][wr0][wk0] =
            *(const uint4*)(W + (size_t)(n0 + wr0) * K + wk0);
        *(uint4*)&Wlds[0][wr1][wk1] =
            *(const uint4*)(W + (size_t)(n0 + wr1) * K + wk1);
    }
    __syncthreads();

    const int nk = K / GT_K;
    for (int ki = 0; ki < nk; ++ki) {
        const int buf = ki & 1;
        // ---- prefetch/stage next chunk into the other buffer ----
        if (ki + 1 < nk) {
            const int nxt = buf ^ 1;
            const int k0  = (ki + 1) * GT_K;
            uint2 av = make_uint2(0u, 0u);
            if (m0 + ar < M)
                av = *(const uint2*)(A + (size_t)(m0 + ar) * K + k0 + akk);
            *(uint2*)&Alds[nxt][ar][akk] = av;
            *(uint4*)&Wlds[nxt][wr0][wk0] =
                *(const uint4*)(W + (size_t)(n0 + wr0) * K + k0 + wk0);
            *(uint4*)&Wlds[nxt][wr1][wk1] =
                *(const uint4*)(W + (size_t)(n0 + wr1) * K + k0 + wk1);
        }

        // ---- fragments from current buffer ----
        FragBF a, b0, b1;
        const int am = wrow + l16;
        #pragma unroll
        for (int r = 0; r < 4; ++r) {
            a.u[r]     = *(const unsigned int*)&Alds[buf][am][kbA + 2 * r];
            a.u[r + 4] = *(const unsigned int*)&Alds[buf][am][8 + kbA + 2 * r];
        }
        const int bn0 = wcol + l16;
        const int bn1 = wcol + 16 + l16;
        #pragma unroll
        for (int r = 0; r < 8; ++r) {
            b0.u[r] = *(const unsigned int*)&Wlds[buf][bn0][kbB + 2 * r];
            b1.u[r] = *(const unsigned int*)&Wlds[buf][bn1][kbB + 2 * r];
        }

        acc0 = __builtin_amdgcn_wmma_f32_16x16x32_bf16(
                   false, a.v, false, b0.v, (short)0, acc0, false, false);
        acc1 = __builtin_amdgcn_wmma_f32_16x16x32_bf16(
                   false, a.v, false, b1.v, (short)0, acc1, false, false);

        __syncthreads();
    }

    // ---- store (C 16x16 f32 layout: lanes 0-15 -> M=r, 16-31 -> M=r+8) ----
    const int cn0 = n0 + wcol + l16;
    const int cn1 = cn0 + 16;
    const float bv0 = bias ? bias[cn0] : 0.0f;
    const float bv1 = bias ? bias[cn1] : 0.0f;
    #pragma unroll
    for (int r = 0; r < 8; ++r) {
        int cm = m0 + wrow + r + ((lane >= 16) ? 8 : 0);
        if (cm < M) {
            float v0 = acc0[r] + bv0;
            float v1 = acc1[r] + bv1;
            if (relu) { v0 = fmaxf(v0, 0.0f); v1 = fmaxf(v1, 0.0f); }
            size_t base = (size_t)cm * N;
            C[base + cn0] = v0;
            C[base + cn1] = v1;
            if (Cbf) {
                Cbf[base + cn0] = f32_to_bf16_bits(v0);
                Cbf[base + cn1] = f32_to_bf16_bits(v1);
            }
        }
    }
}

// ---------------------------------------------------------------------------
// Edge kernel: one wave32 per edge.
//   d1 = |h[row] - h[col]|  (H=256 -> 8 floats / lane via 2x float4)
//   diff_agg[row]  += d1       (per-element f32 atomics, L2-resident)
//   var_accum[row] += var(d1)  (wave shuffle reduction)
//   count[row]     += 1
// ---------------------------------------------------------------------------
__global__ __launch_bounds__(256)
void edge_kernel(const float* __restrict__ h,
                 const long long* __restrict__ ei,
                 float* __restrict__ diff_agg,
                 float* __restrict__ var_accum,
                 float* __restrict__ cnt) {
    const int lane  = threadIdx.x & 31;
    const int wave0 = (blockIdx.x * blockDim.x + threadIdx.x) >> 5;
    const int nw    = (gridDim.x * blockDim.x) >> 5;

    for (int e = wave0; e < EE; e += nw) {
        // prefetch next edge's index pair (emits global_prefetch_b8)
        if (e + nw < EE) {
            __builtin_prefetch(&ei[e + nw], 0, 0);
            __builtin_prefetch(&ei[EE + e + nw], 0, 0);
        }
        const long long row = ei[e];
        const long long col = ei[EE + e];
        const float4* hr = (const float4*)(h + (size_t)row * HH);
        const float4* hc = (const float4*)(h + (size_t)col * HH);
        float* da = diff_agg + (size_t)row * HH;

        float s1 = 0.0f, s2 = 0.0f;
        #pragma unroll
        for (int j = 0; j < 2; ++j) {
            int q = lane * 2 + j;              // float4 slot, 0..63
            float4 a = hr[q];
            float4 b = hc[q];
            float d0 = fabsf(a.x - b.x);
            float d1 = fabsf(a.y - b.y);
            float d2 = fabsf(a.z - b.z);
            float d3 = fabsf(a.w - b.w);
            s1 += d0 + d1 + d2 + d3;
            s2 += d0 * d0 + d1 * d1 + d2 * d2 + d3 * d3;
            atomicAdd(da + q * 4 + 0, d0);
            atomicAdd(da + q * 4 + 1, d1);
            atomicAdd(da + q * 4 + 2, d2);
            atomicAdd(da + q * 4 + 3, d3);
        }
        // wave32 butterfly reduction
        #pragma unroll
        for (int off = 16; off > 0; off >>= 1) {
            s1 += __shfl_xor(s1, off, 32);
            s2 += __shfl_xor(s2, off, 32);
        }
        if (lane == 0) {
            float mu  = s1 * (1.0f / (float)HH);
            float var = (s2 - s1 * mu) * (1.0f / (float)(HH - 1));
            atomicAdd(var_accum + row, var);
            atomicAdd(cnt + row, 1.0f);
        }
    }
}

// ---------------------------------------------------------------------------
// Tau head: per-node tiny MLP -> alpha[N]
// ---------------------------------------------------------------------------
__global__ void tau_kernel(const float* __restrict__ var_accum,
                           const float* __restrict__ cnt,
                           const float* __restrict__ w1,  // [16]
                           const float* __restrict__ b1,  // [16]
                           const float* __restrict__ w2,  // [16]
                           const float* __restrict__ b2,  // [1]
                           float* __restrict__ alpha) {
    int i = blockIdx.x * blockDim.x + threadIdx.x;
    if (i >= NN) return;
    float vm  = var_accum[i] / (cnt[i] + 1e-6f);
    float acc = b2[0];
    #pragma unroll
    for (int j = 0; j < 16; ++j) {
        float t = fmaxf(fmaf(w1[j], vm, b1[j]), 0.0f);
        acc = fmaf(w2[j], t, acc);
    }
    float tau = sigmoidf(acc);
    alpha[i]  = sigmoidf(TAU_THRESH - tau);
}

// ---------------------------------------------------------------------------
// GRU gate math + alpha blend; one thread per (node, element). Updates h in
// place and writes the bf16 shadow used by the next step's GEMMs.
// ---------------------------------------------------------------------------
__global__ void gru_kernel(const float* __restrict__ gi,
                           const float* __restrict__ gh,
                           const float* __restrict__ alpha,
                           float* __restrict__ h,
                           unsigned short* __restrict__ hbf) {
    int t = blockIdx.x * blockDim.x + threadIdx.x;
    if (t >= NN * HH) return;
    int i = t / HH, j = t - i * HH;
    const float* gir = gi + (size_t)i * 3 * HH;
    const float* ghr = gh + (size_t)i * 3 * HH;
    float r = sigmoidf(gir[j]          + ghr[j]);
    float z = sigmoidf(gir[HH + j]     + ghr[HH + j]);
    float n = tanhf(gir[2 * HH + j] + r * ghr[2 * HH + j]);
    float hv = h[t];
    float nh = (1.0f - z) * n + z * hv;
    float a  = alpha[i];
    float out = a * nh + (1.0f - a) * hv;
    h[t]   = out;
    hbf[t] = f32_to_bf16_bits(out);
}

// ---------------------------------------------------------------------------
// Host side
// ---------------------------------------------------------------------------
static inline size_t align256(size_t x) { return (x + 255) & ~(size_t)255; }

extern "C" void kernel_launch(void* const* d_in, const int* in_sizes, int n_in,
                              void* d_out, int out_size, void* d_ws, size_t ws_size,
                              hipStream_t stream) {
    (void)in_sizes; (void)n_in; (void)out_size; (void)ws_size;

    const float*     x      = (const float*)d_in[0];
    const long long* ei     = (const long long*)d_in[1];
    const float*     W_in   = (const float*)d_in[2];
    const float*     b_in   = (const float*)d_in[3];
    const float*     tau_w1 = (const float*)d_in[4];
    const float*     tau_b1 = (const float*)d_in[5];
    const float*     tau_w2 = (const float*)d_in[6];
    const float*     tau_b2 = (const float*)d_in[7];
    const float*     w_ih   = (const float*)d_in[8];
    const float*     w_hh   = (const float*)d_in[9];
    const float*     b_ih   = (const float*)d_in[10];
    const float*     b_hh   = (const float*)d_in[11];
    const float*     W_out  = (const float*)d_in[12];
    const float*     b_out  = (const float*)d_in[13];
    float*           out    = (float*)d_out;

    // ---- workspace carve-up ----
    char* ws = (char*)d_ws;
    size_t off = 0;
    #define CARVE(ptr, type, count) \
        type* ptr = (type*)(ws + off); off = align256(off + (size_t)(count) * sizeof(type));
    CARVE(h,        float,          (size_t)NN * HH)
    CARVE(diff_agg, float,          (size_t)NN * HH)
    CARVE(gi,       float,          (size_t)NN * 3 * HH)
    CARVE(gh,       float,          (size_t)NN * 3 * HH)
    CARVE(var_acc,  float,          NN)
    CARVE(cntb,     float,          NN)
    CARVE(alpha,    float,          NN)
    CARVE(hbf,      unsigned short, (size_t)NN * HH)
    CARVE(dabf,     unsigned short, (size_t)NN * HH)
    CARVE(xbf,      unsigned short, (size_t)NN * IN_DIM)
    CARVE(Winbf,    unsigned short, (size_t)HH * IN_DIM)
    CARVE(wihbf,    unsigned short, (size_t)3 * HH * HH)
    CARVE(whhbf,    unsigned short, (size_t)3 * HH * HH)
    CARVE(Woutbf,   unsigned short, (size_t)OUT_DIM * HH)
    #undef CARVE

    const int T = 256;
    const int mBlocks = (NN + GT_M - 1) / GT_M;   // 313

    // ---- weight / input conversions to bf16 ----
    cvt_f32_bf16_kernel<<<1024, T, 0, stream>>>(x,     xbf,    NN * IN_DIM);
    cvt_f32_bf16_kernel<<<256,  T, 0, stream>>>(W_in,  Winbf,  HH * IN_DIM);
    cvt_f32_bf16_kernel<<<256,  T, 0, stream>>>(w_ih,  wihbf,  3 * HH * HH);
    cvt_f32_bf16_kernel<<<256,  T, 0, stream>>>(w_hh,  whhbf,  3 * HH * HH);
    cvt_f32_bf16_kernel<<<64,   T, 0, stream>>>(W_out, Woutbf, OUT_DIM * HH);

    // ---- h = relu(x @ W_in^T + b_in), plus bf16 shadow ----
    gemm_bf16_wmma_kernel<<<dim3(mBlocks, HH / GT_N), T, 0, stream>>>(
        xbf, Winbf, b_in, h, hbf, NN, HH, IN_DIM, 1);

    for (int step = 0; step < STEPS; ++step) {
        // zero accumulators
        zero_f32_kernel<<<1024, T, 0, stream>>>(diff_agg, NN * HH);
        zero_f32_kernel<<<64,   T, 0, stream>>>(var_acc,  NN);
        zero_f32_kernel<<<64,   T, 0, stream>>>(cntb,     NN);

        // edge gather / scatter-add / variance
        edge_kernel<<<2048, T, 0, stream>>>(h, ei, diff_agg, var_acc, cntb);

        // bf16 shadow of diff_agg for the GEMM
        cvt_f32_bf16_kernel<<<1024, T, 0, stream>>>(diff_agg, dabf, NN * HH);

        // gi = diff_agg @ w_ih^T + b_ih ; gh = h @ w_hh^T + b_hh
        gemm_bf16_wmma_kernel<<<dim3(mBlocks, 3 * HH / GT_N), T, 0, stream>>>(
            dabf, wihbf, b_ih, gi, (unsigned short*)nullptr, NN, 3 * HH, HH, 0);
        gemm_bf16_wmma_kernel<<<dim3(mBlocks, 3 * HH / GT_N), T, 0, stream>>>(
            hbf, whhbf, b_hh, gh, (unsigned short*)nullptr, NN, 3 * HH, HH, 0);

        // alpha from tau head
        tau_kernel<<<(NN + T - 1) / T, T, 0, stream>>>(
            var_acc, cntb, tau_w1, tau_b1, tau_w2, tau_b2, alpha);

        // GRU + blend, in-place h update + bf16 shadow
        gru_kernel<<<(NN * HH + T - 1) / T, T, 0, stream>>>(gi, gh, alpha, h, hbf);
    }

    // ---- out = h @ W_out^T + b_out ----
    gemm_bf16_wmma_kernel<<<dim3(mBlocks, OUT_DIM / GT_N), T, 0, stream>>>(
        hbf, Woutbf, b_out, out, (unsigned short*)nullptr, NN, OUT_DIM, HH, 0);
}
